// EntropicOTQuantileRegression_83391085019888
// MI455X (gfx1250) — matrix operand order repaired and checked by reading, not compiled, not run on hardware
//
#include <hip/hip_runtime.h>

// EntropicOTQuantileRegression fused pairwise-MLP for gfx1250 (MI455X).
//
// ~68.7 GFLOP of 128x128 GEMM over 1M (i,j) pairs vs ~1.2 MB of input ->
// pure matrix-core problem, fully fused: weights in LDS as f16 WMMA B
// fragments, activations staged per-wave in LDS, streaming logsumexp over j.
// Each wave processes M=32 pairs per step so every B fragment loaded from
// LDS feeds TWO v_wmma issues; B fragments are ping-pong double-buffered so
// ds_load latency is covered by in-flight WMMAs. Softplus and the logsumexp
// run in base-2 on raw v_exp_f32/v_log_f32 (no libm fixup tails) so the VALU
// stream stays small and co-executes with the matrix pipe.

#define NPTS   1024
#define FD     64
#define RD     8
#define HID    128
#define EPS    0.1f

typedef __attribute__((ext_vector_type(16))) _Float16 v16h;
typedef __attribute__((ext_vector_type(2)))  _Float16 v2h;
typedef __attribute__((ext_vector_type(8)))  float    v8f;

// Staging row stride (f16 elements): 128 + 8 pad -> 272B row stride, keeps
// 16B alignment for b128 LDS loads while rotating banks across rows.
#define STG_STRIDE 136
#define STG_WAVE   (32 * STG_STRIDE)   // per-wave staging elems (f16), 32 rows

#define LN2f     0.6931471805599453f
#define INVLN2f  1.4426950408889634f

union FragA { v16h v; uint4 q[2]; };
union FragB { v16h v; uint4 q[2]; };
union FragC { v8f  v; float f[8]; };
union Pack8 { _Float16 h[8]; uint4 q; };

__device__ __forceinline__ float fast_exp2(float x) {
#if __has_builtin(__builtin_amdgcn_exp2f)
  return __builtin_amdgcn_exp2f(x);     // bare v_exp_f32 (2^x)
#else
  return exp2f(x);
#endif
}
__device__ __forceinline__ float fast_log2(float x) {
#if __has_builtin(__builtin_amdgcn_logf)
  return __builtin_amdgcn_logf(x);      // bare v_log_f32 (log2 x)
#else
  return log2f(x);
#endif
}

__device__ __forceinline__ float softplusf(float x) {
  // stable softplus in base 2: max(x,0) + ln2*log2(1 + 2^(-|x|/ln2))
  return fmaxf(x, 0.0f) + LN2f * fast_log2(1.0f + fast_exp2(-fabsf(x) * INVLN2f));
}

// ---------------- prep kernels ----------------

// hX[i][h] = sum_k X[i][k] * W0[k][h], k < 64
__global__ void prep_hX(const float* __restrict__ X, const float* __restrict__ W0,
                        float* __restrict__ hX) {
  int idx = blockIdx.x * blockDim.x + threadIdx.x;   // NPTS*HID threads
  int i = idx >> 7, h = idx & 127;
  float acc = 0.0f;
  #pragma unroll 8
  for (int k = 0; k < FD; ++k) acc += X[i * FD + k] * W0[k * HID + h];
  hX[idx] = acc;
}

// hY[j][h] = sum_k Y[j][k] * W0[64+k][h], k < 8
__global__ void prep_hY(const float* __restrict__ Y, const float* __restrict__ W0,
                        float* __restrict__ hY) {
  int idx = blockIdx.x * blockDim.x + threadIdx.x;
  int j = idx >> 7, h = idx & 127;
  float acc = 0.0f;
  #pragma unroll
  for (int k = 0; k < RD; ++k) acc += Y[j * RD + k] * W0[(FD + k) * HID + h];
  hY[idx] = acc;
}

// Pre-swizzle W (128x128, row-major [K][N], f32) into per-lane WMMA B-fragment
// order (f16). B layout (v_wmma_f32_16x16x32_f16, wave32): lanes 0-15 hold
// K {0..7, 16..23}, lanes 16-31 hold K {8..15, 24..31}; N = lane%16.
// Fragment (c=K/32, nb=N/16) stored as 32 lanes x 16 contiguous f16.
__global__ void prep_swz(const float* __restrict__ W, _Float16* __restrict__ Wp) {
  int idx = blockIdx.x * blockDim.x + threadIdx.x;   // 128*128 threads
  int v    = idx & 15;
  int lane = (idx >> 4) & 31;
  int frag = idx >> 9;                               // 0..31 = c*8+nb
  int c  = frag >> 3, nb = frag & 7;
  int K  = 32 * c + 8 * (lane >> 4) + v + 8 * (v >> 3);
  int Nc = 16 * nb + (lane & 15);
  Wp[idx] = (_Float16)W[K * HID + Nc];
}

// ---------------- fused main kernel helpers ----------------

// Load 4 A fragments (K=0..127) from a 16-row f16 staging tile.
// ISA 16-bit A 16x32 layout: lanes 0-15: M=lane, K {0..7, 16..23};
// lanes 16-31: M=lane-16, K {8..15, 24..31}  (+32c per chunk).
__device__ __forceinline__ void load_afrags(const _Float16* stg16, int lane, FragA a[4]) {
  int row  = lane & 15;
  int koff = 8 * (lane >> 4);
  #pragma unroll
  for (int c = 0; c < 4; ++c) {
    int kk = 32 * c + koff;
    a[c].q[0] = *(const uint4*)(stg16 + row * STG_STRIDE + kk);
    a[c].q[1] = *(const uint4*)(stg16 + row * STG_STRIDE + kk + 16);
  }
}

// One 32(pairs) x 128(hidden) layer: 8 N-blocks x 4 K-chunks; each B fragment
// loaded once and used by two WMMAs (row sets 0-15 and 16-31), with a 2-deep
// ping-pong so the next B fragment's ds_loads are in flight behind the current
// WMMA pair. Each N-block's C fragments are softplus'ed + restaged immediately
// (A is already register resident) -> only 2 live C frags.
__device__ __forceinline__ void mlp_layer2(const FragA a0[4], const FragA a1[4],
                                           const _Float16* sW, const float* sb,
                                           int lane, _Float16* stg /*32-row tile*/) {
  const int col0 = lane & 15;
  const int mb   = 8 * (lane >> 4);
  #pragma unroll
  for (int nb = 0; nb < 8; ++nb) {
    float bias = sb[nb * 16 + col0];
    FragC c0, c1;
    c0.v = (v8f){bias, bias, bias, bias, bias, bias, bias, bias};
    c1.v = c0.v;
    FragB bping, bpong;
    {
      const uint4* bp = (const uint4*)(sW + (((0 * 8 + nb) * 32 + lane) << 4));
      bping.q[0] = bp[0];
      bping.q[1] = bp[1];
    }
    #pragma unroll
    for (int cc = 0; cc < 4; ++cc) {
      FragB& cur = (cc & 1) ? bpong : bping;
      FragB& nxt = (cc & 1) ? bping : bpong;
      if (cc < 3) {  // prefetch next K-chunk's B fragment
        const uint4* bp = (const uint4*)(sW + ((((cc + 1) * 8 + nb) * 32 + lane) << 4));
        nxt.q[0] = bp[0];
        nxt.q[1] = bp[1];
      }
      c0.v = __builtin_amdgcn_wmma_f32_16x16x32_f16(
          false, a0[cc].v, false, cur.v, (short)0, c0.v, false, false);
      c1.v = __builtin_amdgcn_wmma_f32_16x16x32_f16(
          false, a1[cc].v, false, cur.v, (short)0, c1.v, false, false);
    }
    // C layout: VGPR r of lane l -> (M = 8*(l/16) + r, N = 16*nb + l%16)
    #pragma unroll
    for (int r = 0; r < 8; ++r) {
      stg[(mb + r) * STG_STRIDE + nb * 16 + col0]        = (_Float16)softplusf(c0.f[r]);
      stg[(16 + mb + r) * STG_STRIDE + nb * 16 + col0]   = (_Float16)softplusf(c1.f[r]);
    }
  }
}

// ---------------- main kernel ----------------
// grid 128 x 256 threads (8 waves). Wave w owns row i = blockIdx*8 + w and
// streams over all 1024 j in chunks of 32 (= 2 WMMA M-tiles).
__global__ __launch_bounds__(256) void eotqr_main(
    const float* __restrict__ U,    const float* __restrict__ Y,
    const float* __restrict__ hX,   const float* __restrict__ hY,
    const _Float16* __restrict__ W1p, const _Float16* __restrict__ W2p,
    const float* __restrict__ b0,   const float* __restrict__ b1,
    const float* __restrict__ b2,   const float* __restrict__ Wout,
    const float* __restrict__ bout, float* __restrict__ out) {
  extern __shared__ char smem[];
  _Float16* sW1   = (_Float16*)smem;                 // 16384 f16
  _Float16* sW2   = sW1 + 16384;                     // 16384 f16
  float*    s_b0  = (float*)(sW2 + 16384);           // 128
  float*    s_b1  = s_b0 + HID;
  float*    s_b2  = s_b1 + HID;
  float*    s_wo  = s_b2 + HID;                      // 128 (f32 fallback)
  float*    s_hX  = s_wo + HID;                      // 8*128
  float*    s_U   = s_hX + 8 * HID;                  // 8*8
  float*    s_hY  = s_U + 64;                        // 32*128
  float*    s_Yc  = s_hY + 32 * HID;                 // 32*8
  _Float16* s_woh = (_Float16*)(s_Yc + 32 * RD);     // 128 f16 Wout
  _Float16* s_stg = s_woh + HID;                     // 8 * STG_WAVE f16

  const int tid  = threadIdx.x;
  const int wave = tid >> 5;
  const int lane = tid & 31;
  const int i0   = blockIdx.x * 8;
  const int i    = i0 + wave;

  // ---- cooperative LDS fill ----
  {
    const uint4* g1 = (const uint4*)W1p;
    const uint4* g2 = (const uint4*)W2p;
    uint4* l1 = (uint4*)sW1;
    uint4* l2 = (uint4*)sW2;
    for (int t = tid; t < 2048; t += 256) { l1[t] = g1[t]; l2[t] = g2[t]; }
    for (int t = tid; t < HID; t += 256) {
      float w = Wout[t];
      s_b0[t] = b0[t]; s_b1[t] = b1[t]; s_b2[t] = b2[t];
      s_wo[t] = w;     s_woh[t] = (_Float16)w;
    }
    for (int t = tid; t < 8 * HID; t += 256) s_hX[t] = hX[i0 * HID + t];
    if (tid < 64) s_U[tid] = U[i0 * RD + tid];
  }
  __syncthreads();

  const float bo = bout[0];
  _Float16* stg = s_stg + wave * STG_WAVE;
  const float* hXr = s_hX + wave * HID;

  // streaming logsumexp over j in the log2 domain:
  // track m = max(slack/(EPS*ln2)), s = sum 2^(t-m)
  float run_m = -INFINITY;
  float run_s = 0.0f;

  for (int jc = 0; jc < 32; ++jc) {
    __syncthreads();   // previous chunk fully consumed before overwrite
    {  // load hY chunk (32x128 f32) + Y chunk (32x8) cooperatively
      const float4* g = (const float4*)(hY + jc * 32 * HID);
      float4* l = (float4*)s_hY;
      for (int t = tid; t < 1024; t += 256) l[t] = g[t];
      for (int t = tid; t < 32 * RD; t += 256) s_Yc[t] = Y[jc * 32 * RD + t];
      if (jc + 1 < 32) __builtin_prefetch(hY + (jc + 1) * 32 * HID, 0, 1);
    }
    __syncthreads();

    // ---- layer 0: h0 = softplus(hX[i] + hY[j] + b0) -> f16 staging ----
    // lane owns pair row p = lane (all 128 hidden), packed b128 stores.
    {
      const float* hYr = s_hY + lane * HID;
      _Float16* srow = stg + lane * STG_STRIDE;
      #pragma unroll 2
      for (int k0 = 0; k0 < HID; k0 += 8) {
        Pack8 pk;
        #pragma unroll
        for (int e = 0; e < 8; ++e) {
          float x = hXr[k0 + e] + hYr[k0 + e] + s_b0[k0 + e];
          pk.h[e] = (_Float16)softplusf(x);
        }
        *(uint4*)(srow + k0) = pk.q;
      }
    }

    FragA a0[4], a1[4];

    // ---- layer 1 ----
    load_afrags(stg, lane, a0);
    load_afrags(stg + 16 * STG_STRIDE, lane, a1);
    mlp_layer2(a0, a1, sW1, s_b1, lane, stg);

    // ---- layer 2 ----
    load_afrags(stg, lane, a0);
    load_afrags(stg + 16 * STG_STRIDE, lane, a1);
    mlp_layer2(a0, a1, sW2, s_b2, lane, stg);

    // ---- psi = h2 . Wout + bout ; lane owns pair p = lane ----
    float psi = bo;
    {
#if __has_builtin(__builtin_amdgcn_fdot2)
      const v2h* hr2 = (const v2h*)(stg + lane * STG_STRIDE);
      const v2h* wo2 = (const v2h*)s_woh;
      #pragma unroll 8
      for (int k = 0; k < HID / 2; ++k)
        psi = __builtin_amdgcn_fdot2(hr2[k], wo2[k], psi, false);
#else
      const _Float16* hr = stg + lane * STG_STRIDE;
      #pragma unroll 8
      for (int k = 0; k < HID; ++k) psi += (float)hr[k] * s_wo[k];
#endif
    }

    // ---- slack = U[i].Y[j] - psi ; stream into base-2 logsumexp ----
    float cost = 0.0f;
    {
      const float* Ur = s_U + wave * RD;
      const float* Yr = s_Yc + lane * RD;
      #pragma unroll
      for (int r = 0; r < RD; ++r) cost += Ur[r] * Yr[r];
    }
    float t  = (cost - psi) * (INVLN2f / EPS);      // slack/(EPS*ln2)
    float nm = fmaxf(run_m, t);
    run_s = run_s * fast_exp2(run_m - nm) + fast_exp2(t - nm);
    run_m = nm;
  }

  // merge (m,s) across all 32 lanes (each lane owned distinct j slots)
  #pragma unroll
  for (int m = 1; m < 32; m <<= 1) {
    float om = __shfl_xor(run_m, m);
    float os = __shfl_xor(run_s, m);
    float nm = fmaxf(run_m, om);
    run_s = run_s * fast_exp2(run_m - nm) + os * fast_exp2(om - nm);
    run_m = nm;
  }
  // psi_est = EPS*ln2*(m + log2(s) - log2(1024))
  if (lane == 0)
    out[i] = (EPS * LN2f) * (run_m + fast_log2(run_s) - 10.0f);
}

// ---------------- launch ----------------
extern "C" void kernel_launch(void* const* d_in, const int* in_sizes, int n_in,
                              void* d_out, int out_size, void* d_ws, size_t ws_size,
                              hipStream_t stream) {
  (void)in_sizes; (void)n_in; (void)out_size; (void)ws_size;
  const float* X    = (const float*)d_in[0];
  const float* U    = (const float*)d_in[1];
  const float* Y    = (const float*)d_in[2];
  const float* W0   = (const float*)d_in[3];
  const float* b0   = (const float*)d_in[4];
  const float* W1   = (const float*)d_in[5];
  const float* b1   = (const float*)d_in[6];
  const float* W2   = (const float*)d_in[7];
  const float* b2   = (const float*)d_in[8];
  const float* Wout = (const float*)d_in[9];
  const float* bout = (const float*)d_in[10];
  float* out = (float*)d_out;

  char* ws = (char*)d_ws;
  float*    hX  = (float*)ws;     ws += (size_t)NPTS * HID * 4;
  float*    hY  = (float*)ws;     ws += (size_t)NPTS * HID * 4;
  _Float16* W1p = (_Float16*)ws;  ws += (size_t)HID * HID * 2;
  _Float16* W2p = (_Float16*)ws;  ws += (size_t)HID * HID * 2;

  prep_hX<<<(NPTS * HID) / 256, 256, 0, stream>>>(X, W0, hX);
  prep_hY<<<(NPTS * HID) / 256, 256, 0, stream>>>(Y, W0, hY);
  prep_swz<<<(HID * HID) / 256, 256, 0, stream>>>(W1, W1p);
  prep_swz<<<(HID * HID) / 256, 256, 0, stream>>>(W2, W2p);

  // dynamic LDS: W1+W2 f16 (64KB) + biases/Wout + hX tile + U + hY chunk
  // (32x128 f32) + Y chunk + f16 Wout + 8 per-wave 32-row staging = 159232 B
  size_t shmem = 2 * 16384 * 2                       // sW1, sW2
               + 4 * HID * 4                          // b0,b1,b2,Wout(f32)
               + 8 * HID * 4                          // hX tile
               + 64 * 4                               // U tile
               + 32 * HID * 4                         // hY chunk
               + 32 * RD * 4                          // Y chunk
               + HID * 2                              // Wout f16
               + 8 * STG_WAVE * 2;                    // wave staging
  eotqr_main<<<NPTS / 8, 256, shmem, stream>>>(U, Y, hX, hY, W1p, W2p,
                                               b0, b1, b2, Wout, bout, out);
}
